// RPN_40097814675727
// MI455X (gfx1250) — compile-verified, matrix operand order checked by reading, not run on hardware
//
#include <hip/hip_runtime.h>
#include <hip/hip_bf16.h>
#include <cfloat>

typedef __attribute__((ext_vector_type(16))) __bf16 v16bf;
typedef __attribute__((ext_vector_type(8)))  __bf16 v8bf;
typedef __attribute__((ext_vector_type(8)))  float  v8f;
typedef unsigned int u32x4 __attribute__((ext_vector_type(4)));
typedef int          i32x8 __attribute__((ext_vector_type(8)));
typedef int          i32x4 __attribute__((ext_vector_type(4)));

#define BATCH   2
#define DIN     1024
#define DMID    512
#define HH      50
#define WWID    84
#define HP2     60               // padded rows: max touched = 48+7+2 = 57
#define WP2     100              // padded cols: max touched = 80+15+2 = 97
#define NPIX    (HH*WWID)        // 4200
#define KCONV   (9*DIN)          // 9216
#define NA      9
#define KPROP   (NPIX*NA)        // 37800
#define PRE_N   6000
#define POST_N  300
#define CAP     6144
#define NMS_TH  0.7f
#define MHEAD   64               // 18 cls + 36 bbox + 10 pad
#define NPAD    4224             // 33*128
#define LDSS    40               // LDS row stride (bf16): 80B = 64B data + 16B TDM pad

// ---------------- TDM: tensor_load_to_lds with 3D tile + LDS row padding ----------------
// Tile: X = tile0 elems (contiguous, 2B each), Y = tile1 (stride s0 elems), Z = tile2 (stride s1).
// pad_interval=3 (16 DWORDs = 64B), pad_amount=3 (4 DWORDs = 16B) -> LDS row stride 80B (= LDSS).
__device__ __forceinline__ void tdm_load(unsigned lds_addr, unsigned long long gaddr,
                                         int tile0, int tile1, int tile2,
                                         unsigned long long s0, unsigned long long s1) {
  u32x4 g0;
  g0[0] = 1u;                                            // count=1 (valid descriptor)
  g0[1] = lds_addr;                                      // LDS byte address
  g0[2] = (unsigned)gaddr;                               // global_addr[31:0]
  g0[3] = (unsigned)(gaddr >> 32) | (2u << 30);          // global_addr[56:32] | type=2
  const unsigned TD = 1u << 20;                          // huge tensor dims: no OOB clamping
  i32x8 g1;
  g1[0] = (int)((1u << 16) | (1u << 20) | (3u << 22) | (3u << 25)); // 2B elems, pad 16B/64B
  g1[1] = (int)((TD & 0xffffu) << 16);                   // abar=0 | tensor_dim0[15:0]
  g1[2] = (int)((TD >> 16) | ((TD & 0xffffu) << 16));    // td0[31:16] | td1[15:0]
  g1[3] = (int)((TD >> 16) | ((unsigned)tile0 << 16));   // td1[31:16] | tile_dim0
  g1[4] = (int)((unsigned)tile1 | ((unsigned)tile2 << 16));
  g1[5] = (int)(unsigned)s0;                             // dim0_stride[31:0]
  g1[6] = (int)(((unsigned)(s0 >> 32) & 0xffffu) | (((unsigned)s1 & 0xffffu) << 16));
  g1[7] = (int)(unsigned)(s1 >> 16);                     // dim1_stride[47:16]
  i32x4 g2; g2[0] = (int)TD; g2[1] = 0; g2[2] = 0; g2[3] = 0;   // tensor_dim2; rest 0
  i32x4 g3 = {0, 0, 0, 0};
#if defined(__clang_major__) && (__clang_major__ >= 23)
  i32x8 g4 = {0, 0, 0, 0, 0, 0, 0, 0};
  __builtin_amdgcn_tensor_load_to_lds(g0, g1, g2, g3, g4, 0);
#else
  __builtin_amdgcn_tensor_load_to_lds(g0, g1, g2, g3, 0);
#endif
}

// ---------------- fragment load (matches ISA 16-bit A/B VGPR layout) ----------------
__device__ __forceinline__ v16bf load_frag(const __bf16* tileBase, int lane) {
  int r  = lane & 15;
  int kg = (lane >> 4) << 3;       // lanes 0-15: K {0..7,16..23}; lanes 16-31: {8..15,24..31}
  const __bf16* p = tileBase + r * LDSS + kg;
  v8bf lo = *(const v8bf*)(p);
  v8bf hi = *(const v8bf*)(p + 16);
  v16bf o;
#pragma unroll
  for (int i = 0; i < 8; ++i) { o[i] = lo[i]; o[i + 8] = hi[i]; }
  return o;
}

// ---------------- init: zero padded acts (halo), hist; candScores = -inf ----------------
__global__ void k_init(unsigned int* actU32, int* hist, float* candS) {
  size_t n = (size_t)BATCH * HP2 * WP2 * DIN / 2;      // u32 count of bf16 buffer
  size_t i = (size_t)blockIdx.x * blockDim.x + threadIdx.x;
  size_t stride = (size_t)gridDim.x * blockDim.x;
  for (size_t j = i; j < n; j += stride) actU32[j] = 0u;
  if (i < BATCH * 1024) hist[i] = 0;
  if (i < BATCH * CAP)  candS[i] = -FLT_MAX;
}

// ---------------- weight packs ----------------
__global__ void k_pack_conv(const float* __restrict__ Wc, __bf16* __restrict__ Wpack) {
  size_t t = (size_t)blockIdx.x * blockDim.x + threadIdx.x;
  if (t >= (size_t)DMID * KCONV) return;
  int oc = (int)(t / KCONV), k = (int)(t % KCONV);
  int pos = k >> 10, ic = k & 1023;                    // k = pos*1024 + ic
  Wpack[t] = (__bf16)Wc[((size_t)oc * DIN + ic) * 9 + pos];
}

__global__ void k_pack_head(const float* __restrict__ Wcls, const float* __restrict__ Wbbox,
                            __bf16* __restrict__ Whead) {
  int t = blockIdx.x * blockDim.x + threadIdx.x;
  if (t >= MHEAD * DMID) return;
  int row = t >> 9, k = t & 511;
  float v = 0.f;
  if (row < 18)      v = Wcls[row * DMID + k];
  else if (row < 54) v = Wbbox[(row - 18) * DMID + k];
  Whead[t] = (__bf16)v;
}

// ---------------- NCHW f32 -> padded channel-last bf16 (LDS transpose) ----------------
__global__ __launch_bounds__(256) void k_actpad(const float* __restrict__ feat,
                                                __bf16* __restrict__ actCL) {
  __shared__ float tile[16][17];
  int h = blockIdx.x, icg = blockIdx.y, b = blockIdx.z;
  int t = threadIdx.x;
  int icL = t >> 4, wL = t & 15;
  for (int wb = 0; wb < 6; ++wb) {
    int w = wb * 16 + wL;
    float v = 0.f;
    if (w < WWID) v = feat[(((size_t)b * DIN + icg * 16 + icL) * HH + h) * WWID + w];
    tile[icL][wL] = v;
    __syncthreads();
    int wL2 = t >> 4, icL2 = t & 15;
    int w2 = wb * 16 + wL2;
    if (w2 < WWID)
      actCL[(((size_t)b * HP2 + (h + 1)) * WP2 + (w2 + 1)) * DIN + icg * 16 + icL2] =
          (__bf16)tile[icL2][wL2];
    __syncthreads();
  }
}

// ---------------- 3x3 conv as implicit-im2col WMMA GEMM, TDM double-buffered ----------------
// C[512, 4200] = Wpack[512, 9216] x im2col(act)[9216, 4200], per batch.
// Block: 256 thr (8 waves, 4Mx2N). Wave 0 drives TDM; all waves do ds_load_b128 + WMMA.
__global__ __launch_bounds__(256) void k_conv3x3(const __bf16* __restrict__ Wpack,
                                                 const __bf16* __restrict__ actCL,
                                                 const float* __restrict__ bconv,
                                                 __bf16* __restrict__ xCL) {
  __shared__ __attribute__((aligned(128))) __bf16 Abuf[2][128 * LDSS];
  __shared__ __attribute__((aligned(128))) __bf16 Bbuf[2][128 * LDSS];
  int tid = threadIdx.x, lane = tid & 31, wave = tid >> 5;
  int waveM = wave & 3, waveN = wave >> 2;

  int bx = blockIdx.x;
  int mt = bx & 3; bx >>= 2;
  int wt = bx % 6; bx /= 6;
  int ht = bx % 7; int b = bx / 7;
  int m0 = mt * 128, h0 = ht * 8, w0 = wt * 16;

  v8f acc[2][4] = {};

  unsigned long long wAddr = (unsigned long long)(uintptr_t)Wpack;
  unsigned long long aAddr = (unsigned long long)(uintptr_t)actCL;
  unsigned ldsA[2] = {(unsigned)(uintptr_t)&Abuf[0][0], (unsigned)(uintptr_t)&Abuf[1][0]};
  unsigned ldsB[2] = {(unsigned)(uintptr_t)&Bbuf[0][0], (unsigned)(uintptr_t)&Bbuf[1][0]};

  auto issue = [&](int s, int par) {
    int pos = s >> 5, c = s & 31;
    int kh = pos / 3, kw = pos % 3;
    // A: 2D (z=1): 128 rows (oc) x 32 k, row stride KCONV elems
    tdm_load(ldsA[par],
             wAddr + ((size_t)m0 * KCONV + (size_t)pos * DIN + c * 32) * 2,
             32, 128, 1, (unsigned long long)KCONV, 0ull);
    // B: 3D: 32 ic x 16 w (stride DIN) x 8 h (stride WP2*DIN)
    tdm_load(ldsB[par],
             aAddr + ((((size_t)b * HP2 + (h0 + kh)) * WP2 + (w0 + kw)) * DIN + c * 32) * 2,
             32, 16, 8, (unsigned long long)DIN, (unsigned long long)WP2 * DIN);
  };

  // prologue: stage step 0
  if (wave == 0) {
    issue(0, 0);
    __builtin_amdgcn_s_wait_tensorcnt(0);
  }
  __syncthreads();

  for (int s = 0; s < 288; ++s) {
    int par = s & 1;
    if (wave == 0 && s + 1 < 288) issue(s + 1, par ^ 1);   // DMA next while computing

    v16bf af[2], bfr[4];
#pragma unroll
    for (int i = 0; i < 2; ++i)
      af[i] = load_frag(&Abuf[par][(waveM * 32 + i * 16) * LDSS], lane);
#pragma unroll
    for (int j = 0; j < 4; ++j)
      bfr[j] = load_frag(&Bbuf[par][(waveN * 64 + j * 16) * LDSS], lane);
#pragma unroll
    for (int i = 0; i < 2; ++i)
#pragma unroll
      for (int j = 0; j < 4; ++j)
        acc[i][j] = __builtin_amdgcn_wmma_f32_16x16x32_bf16(
            false, af[i], false, bfr[j], (short)0, acc[i][j], false, false);

    __syncthreads();                                       // all done reading buf[par]
    if (wave == 0 && s + 1 < 288) __builtin_amdgcn_s_wait_tensorcnt(0);
    __syncthreads();                                       // buf[par^1] published
  }

  // epilogue: +bias, ReLU, bf16, channel-last store (oc contiguous over VGPR index r)
  int laneLo = lane & 15, laneHi = lane >> 4;
#pragma unroll
  for (int i = 0; i < 2; ++i) {
    int ocb = m0 + waveM * 32 + i * 16 + laneHi * 8;
    float bias[8];
#pragma unroll
    for (int r = 0; r < 8; ++r) bias[r] = bconv[ocb + r];
#pragma unroll
    for (int j = 0; j < 4; ++j) {
      int n = waveN * 64 + j * 16 + laneLo;
      int hh = h0 + (n >> 4), ww = w0 + (n & 15);
      if (hh < HH && ww < WWID) {
        int p = hh * WWID + ww;
        v8bf outv;
#pragma unroll
        for (int r = 0; r < 8; ++r) {
          float v = acc[i][j][r] + bias[r];
          outv[r] = (__bf16)(v > 0.f ? v : 0.f);
        }
        *(v8bf*)&xCL[((size_t)b * NPIX + p) * DMID + ocb] = outv;
      }
    }
  }
}

// ---------------- 1x1 heads (cls+bbox fused): [64,4200] = Whead[64,512] x xCL^T ----------------
__global__ __launch_bounds__(256) void k_head(const __bf16* __restrict__ Whead,
                                              const __bf16* __restrict__ xCL,
                                              float* __restrict__ headOut) {
  __shared__ __bf16 Alds[64 * LDSS];
  __shared__ __bf16 Blds[128 * LDSS];
  int tid = threadIdx.x, lane = tid & 31, wave = tid >> 5;
  int waveM = wave & 1, waveN = wave >> 1;             // 2(M) x 4(N)
  int nt = blockIdx.x, b = blockIdx.y;
  int n0 = nt * 128;
  v8f acc[2][2] = {};

  int aRow = tid >> 2, aQ = tid & 3;                   // 64 rows x 4 quarters (8 bf16)
  int bRow = tid >> 1, bHalf = tid & 1;
  int pB = n0 + bRow;
  bool pIn = pB < NPIX;

  for (int ks = 0; ks < 16; ++ks) {
    int k0 = ks * 32;
    v8bf av = *(const v8bf*)(Whead + (size_t)aRow * DMID + k0 + aQ * 8);
    v8bf b0 = {}, b1 = {};
    if (pIn) {
      const __bf16* src = xCL + ((size_t)b * NPIX + pB) * DMID + k0 + bHalf * 16;
      b0 = *(const v8bf*)src; b1 = *(const v8bf*)(src + 8);
    }
    *(v8bf*)&Alds[aRow * LDSS + aQ * 8] = av;
    __bf16* bDst = &Blds[bRow * LDSS + bHalf * 16];
    *(v8bf*)bDst = b0; *(v8bf*)(bDst + 8) = b1;
    __syncthreads();

    v16bf af[2], bfr[2];
#pragma unroll
    for (int i = 0; i < 2; ++i) af[i]  = load_frag(&Alds[(waveM * 32 + i * 16) * LDSS], lane);
#pragma unroll
    for (int j = 0; j < 2; ++j) bfr[j] = load_frag(&Blds[(waveN * 32 + j * 16) * LDSS], lane);
#pragma unroll
    for (int i = 0; i < 2; ++i)
#pragma unroll
      for (int j = 0; j < 2; ++j)
        acc[i][j] = __builtin_amdgcn_wmma_f32_16x16x32_bf16(
            false, af[i], false, bfr[j], (short)0, acc[i][j], false, false);
    __syncthreads();
  }

  int laneLo = lane & 15, laneHi = lane >> 4;
#pragma unroll
  for (int i = 0; i < 2; ++i) {
    int ocb = waveM * 32 + i * 16 + laneHi * 8;
#pragma unroll
    for (int j = 0; j < 2; ++j) {
      int p = n0 + waveN * 32 + j * 16 + laneLo;
      if (p < NPIX)
        *(v8f*)(headOut + ((size_t)b * NPAD + p) * MHEAD + ocb) = acc[i][j];
    }
  }
}

// ---------------- softmax + bbox_transform_inv + clip ----------------
__global__ void k_prop(const float* __restrict__ headOut,
                       const float* __restrict__ bcls, const float* __restrict__ bbbox,
                       const float* __restrict__ im_info,
                       float* __restrict__ scores, float* __restrict__ props) {
  int idx = blockIdx.x * blockDim.x + threadIdx.x;
  if (idx >= BATCH * NPIX) return;
  int b = idx / NPIX, p = idx % NPIX;
  int h = p / WWID, w = p % WWID;
  const float* raw = headOut + ((size_t)b * NPAD + p) * MHEAD;
  float im_h = im_info[b * 3 + 0], im_w = im_info[b * 3 + 1];
  const float wsA[9] = {23.f,23.f,23.f,16.f,16.f,16.f,11.f,11.f,11.f};
  const float hsA[9] = {12.f,12.f,12.f,16.f,16.f,16.f,22.f,22.f,22.f};
  const float scA[3] = {8.f,16.f,32.f};
  float sx = w * 16.f, sy = h * 16.f;
#pragma unroll
  for (int a = 0; a < 9; ++a) {
    float s0 = raw[a] + bcls[a];
    float s1 = raw[NA + a] + bcls[NA + a];
    float fg = 1.f / (1.f + expf(s0 - s1));
    float dx = raw[18 + 4 * a + 0] + bbbox[4 * a + 0];
    float dy = raw[18 + 4 * a + 1] + bbbox[4 * a + 1];
    float dw = raw[18 + 4 * a + 2] + bbbox[4 * a + 2];
    float dh = raw[18 + 4 * a + 3] + bbbox[4 * a + 3];
    float AW = wsA[a] * scA[a % 3], AHt = hsA[a] * scA[a % 3];
    float cx = sx + 8.f, cy = sy + 8.f;                // anchor centers collapse to sx+8
    float px = dx * AW + cx, py = dy * AHt + cy;
    float pw = expf(dw) * AW, ph = expf(dh) * AHt;
    float x1 = fminf(fmaxf(px - 0.5f * pw, 0.f), im_w - 1.f);
    float y1 = fminf(fmaxf(py - 0.5f * ph, 0.f), im_h - 1.f);
    float x2 = fminf(fmaxf(px + 0.5f * pw, 0.f), im_w - 1.f);
    float y2 = fminf(fmaxf(py + 0.5f * ph, 0.f), im_h - 1.f);
    size_t k = (size_t)b * KPROP + (size_t)p * NA + a;
    scores[k] = fg;
    ((float4*)props)[k] = make_float4(x1, y1, x2, y2);
  }
}

// ---------------- score histogram for approximate top-PRE_N threshold ----------------
__global__ void k_hist(const float* __restrict__ scores, int* __restrict__ hist) {
  int i = blockIdx.x * blockDim.x + threadIdx.x;
  if (i >= BATCH * KPROP) return;
  int b = i / KPROP;
  int bin = (int)(scores[i] * 1024.f);
  bin = bin < 0 ? 0 : (bin > 1023 ? 1023 : bin);
  atomicAdd(&hist[b * 1024 + bin], 1);
}

// ---------------- deterministic threshold + compaction ----------------
__global__ __launch_bounds__(1024) void k_compact(const float* __restrict__ scores,
                                                  const float* __restrict__ props,
                                                  const int* __restrict__ hist,
                                                  float* __restrict__ candS,
                                                  float* __restrict__ candB) {
  int b = blockIdx.x, tid = threadIdx.x;
  __shared__ int s_thr;
  __shared__ int cnt[1024];
  __shared__ int off[1025];
  if (tid == 0) {
    int cum = 0, thr = 0;
    for (int i = 1023; i >= 0; --i) { cum += hist[b * 1024 + i]; if (cum >= PRE_N) { thr = i; break; } }
    s_thr = thr;
  }
  __syncthreads();
  int thr = s_thr;
  int c = 0;
  for (int i = tid; i < KPROP; i += 1024) {
    int bin = (int)(scores[(size_t)b * KPROP + i] * 1024.f);
    bin = bin < 0 ? 0 : (bin > 1023 ? 1023 : bin);
    if (bin >= thr) ++c;
  }
  cnt[tid] = c;
  __syncthreads();
  if (tid == 0) { int s = 0; for (int i = 0; i < 1024; ++i) { off[i] = s; s += cnt[i]; } off[1024] = s; }
  __syncthreads();
  int o = off[tid];
  for (int i = tid; i < KPROP; i += 1024) {
    float s = scores[(size_t)b * KPROP + i];
    int bin = (int)(s * 1024.f);
    bin = bin < 0 ? 0 : (bin > 1023 ? 1023 : bin);
    if (bin >= thr) {
      if (o < CAP) {
        candS[b * CAP + o] = s;
        ((float4*)candB)[b * CAP + o] = ((const float4*)props)[(size_t)b * KPROP + i];
      }
      ++o;
    }
  }
}

// ---------------- greedy NMS, one workgroup per batch, scores in LDS ----------------
__global__ __launch_bounds__(1024) void k_nms(const float* __restrict__ candS,
                                              const float* __restrict__ candB,
                                              float* __restrict__ rois) {
  int b = blockIdx.x, tid = threadIdx.x;
  __shared__ float s_sc[CAP];
  __shared__ float red_v[1024];
  __shared__ int   red_i[1024];
  __shared__ float s_win[5];
  for (int i = tid; i < CAP; i += 1024) s_sc[i] = candS[b * CAP + i];
  __syncthreads();
  const float4* boxes = (const float4*)candB + (size_t)b * CAP;

  for (int r = 0; r < POST_N; ++r) {
    float bv = -FLT_MAX; int bi = 0;
    for (int i = tid; i < CAP; i += 1024) {
      float v = s_sc[i];
      if (v > bv) { bv = v; bi = i; }
    }
    red_v[tid] = bv; red_i[tid] = bi;
    __syncthreads();
    for (int s = 512; s > 0; s >>= 1) {
      if (tid < s) {
        float ov = red_v[tid + s]; int oi = red_i[tid + s];
        if (ov > red_v[tid] || (ov == red_v[tid] && oi < red_i[tid])) { red_v[tid] = ov; red_i[tid] = oi; }
      }
      __syncthreads();
    }
    if (tid == 0) {
      int wi = red_i[0]; float wv = red_v[0];
      float valid = (wv > -1e30f) ? 1.f : 0.f;
      float4 wb = boxes[wi];
      s_win[0] = wb.x; s_win[1] = wb.y; s_win[2] = wb.z; s_win[3] = wb.w; s_win[4] = valid;
      float* o = rois + ((size_t)b * POST_N + r) * 5;
      o[0] = (float)b;
      o[1] = wb.x * valid; o[2] = wb.y * valid; o[3] = wb.z * valid; o[4] = wb.w * valid;
      s_sc[wi] = -FLT_MAX;
    }
    __syncthreads();
    if (s_win[4] > 0.5f) {
      float wx1 = s_win[0], wy1 = s_win[1], wx2 = s_win[2], wy2 = s_win[3];
      float wa = (wx2 - wx1 + 1.f) * (wy2 - wy1 + 1.f);
      for (int i = tid; i < CAP; i += 1024) {
        if (s_sc[i] > -1e30f) {
          float4 bb = boxes[i];
          float ix1 = fmaxf(wx1, bb.x), iy1 = fmaxf(wy1, bb.y);
          float ix2 = fminf(wx2, bb.z), iy2 = fminf(wy2, bb.w);
          float inter = fmaxf(ix2 - ix1 + 1.f, 0.f) * fmaxf(iy2 - iy1 + 1.f, 0.f);
          float oa = (bb.z - bb.x + 1.f) * (bb.w - bb.y + 1.f);
          if (inter / (wa + oa - inter) > NMS_TH) s_sc[i] = -FLT_MAX;
        }
      }
    }
    __syncthreads();
  }
}

// ---------------- host launcher ----------------
static inline size_t alignUp(size_t x) { return (x + 255) & ~(size_t)255; }

extern "C" void kernel_launch(void* const* d_in, const int* in_sizes, int n_in,
                              void* d_out, int out_size, void* d_ws, size_t ws_size,
                              hipStream_t stream) {
  const float* base_feat = (const float*)d_in[0];
  const float* im_info   = (const float*)d_in[1];
  const float* W_conv    = (const float*)d_in[2];
  const float* b_conv    = (const float*)d_in[3];
  const float* W_cls     = (const float*)d_in[4];
  const float* b_cls     = (const float*)d_in[5];
  const float* W_bbox    = (const float*)d_in[6];
  const float* b_bbox    = (const float*)d_in[7];
  float* rois = (float*)d_out;

  char* ws = (char*)d_ws;
  size_t off = 0;
  __bf16* actCL   = (__bf16*)(ws + off); off = alignUp(off + (size_t)BATCH * HP2 * WP2 * DIN * 2);
  __bf16* Wpack   = (__bf16*)(ws + off); off = alignUp(off + (size_t)DMID * KCONV * 2);
  __bf16* xCL     = (__bf16*)(ws + off); off = alignUp(off + (size_t)BATCH * NPIX * DMID * 2);
  __bf16* Whead   = (__bf16*)(ws + off); off = alignUp(off + (size_t)MHEAD * DMID * 2);
  float*  headOut = (float*) (ws + off); off = alignUp(off + (size_t)BATCH * NPAD * MHEAD * 4);
  float*  scores  = (float*) (ws + off); off = alignUp(off + (size_t)BATCH * KPROP * 4);
  float*  props   = (float*) (ws + off); off = alignUp(off + (size_t)BATCH * KPROP * 16);
  int*    hist    = (int*)   (ws + off); off = alignUp(off + (size_t)BATCH * 1024 * 4);
  float*  candS   = (float*) (ws + off); off = alignUp(off + (size_t)BATCH * CAP * 4);
  float*  candB   = (float*) (ws + off); off = alignUp(off + (size_t)BATCH * CAP * 16);
  if (ws_size < off) return;  // workspace too small; avoid OOB

  {
    size_t n = (size_t)BATCH * HP2 * WP2 * DIN / 2;
    k_init<<<(unsigned)((n + 1023) / 1024), 1024, 0, stream>>>((unsigned int*)actCL, hist, candS);
  }
  {
    size_t n = (size_t)DMID * KCONV;
    k_pack_conv<<<(unsigned)((n + 1023) / 1024), 1024, 0, stream>>>(W_conv, Wpack);
  }
  k_pack_head<<<(MHEAD * DMID + 1023) / 1024, 1024, 0, stream>>>(W_cls, W_bbox, Whead);
  k_actpad<<<dim3(HH, DIN / 16, BATCH), 256, 0, stream>>>(base_feat, actCL);
  k_conv3x3<<<4 * 6 * 7 * BATCH, 256, 0, stream>>>(Wpack, actCL, b_conv, xCL);
  k_head<<<dim3(NPAD / 128, BATCH), 256, 0, stream>>>(Whead, xCL, headOut);
  k_prop<<<(BATCH * NPIX + 255) / 256, 256, 0, stream>>>(headOut, b_cls, b_bbox, im_info,
                                                         scores, props);
  k_hist<<<(BATCH * KPROP + 255) / 256, 256, 0, stream>>>(scores, hist);
  k_compact<<<BATCH, 1024, 0, stream>>>(scores, props, hist, candS, candB);
  k_nms<<<BATCH, 1024, 0, stream>>>(candS, candB, rois);
}
